// SkillMatchingModel_60765197304106
// MI455X (gfx1250) — matrix operand order repaired and checked by reading, not compiled, not run on hardware
//
#include <hip/hip_runtime.h>
#include <cmath>

// ---------------------------------------------------------------------------
// GAT-style GNN for MI455X (gfx1250, wave32).
//   * dense GEMMs (edge encoder E x 64 x 32, node transform N x 128 x 128)
//     run on v_wmma_f32_16x16x32_f16 with B staged as f16 in LDS (transposed)
//   * node-transform weight tile is DMA'd global->LDS by the Tensor Data
//     Mover (tensor_load_to_lds + s_wait_tensorcnt)
//   * al_src / al_dst / al_e are algebraically folded into tiny GEMVs
//     (W (.) a), eliminating the (E+N) x 32 x 128 "ee" GEMM entirely
//   * segment softmax + aggregation via native float atomics
// ---------------------------------------------------------------------------

typedef __attribute__((ext_vector_type(16))) _Float16 v16h;
typedef __attribute__((ext_vector_type(8)))  float    v8f;
typedef __attribute__((ext_vector_type(4)))  unsigned int u32x4;
typedef __attribute__((ext_vector_type(8)))  int      i32x8;
typedef __attribute__((ext_vector_type(4)))  int      i32x4;

#define DM   128   // node feature dim (in/hid/out)
#define HDIM 32    // edge-encoder output dim / per-head dim (layer 0)
#define EDIM 64    // raw edge feature dim

// ---------------------------------------------------------------------- utils

__device__ inline void atomicMaxFloat(float* addr, float val) {
  // monotone float max via int/uint atomics (buffer initialized to -inf)
  if (val >= 0.0f) atomicMax((int*)addr, __float_as_int(val));
  else             atomicMin((unsigned int*)addr, __float_as_uint(val));
}

__global__ void fill_kernel(float* __restrict__ p, float v, int n) {
  int t = blockIdx.x * blockDim.x + threadIdx.x;
  if (t < n) p[t] = v;
}

// out[k*H+h] = sum_c W[k, h*C + c] * a[h, c]      (fold big matrix with attn vec)
__global__ void fold_kernel(const float* __restrict__ W, const float* __restrict__ a,
                            float* __restrict__ out, int K, int H, int C) {
  int t = blockIdx.x * blockDim.x + threadIdx.x;
  if (t >= K * H) return;
  int k = t / H, h = t % H;
  float s = 0.0f;
  for (int c = 0; c < C; ++c) s += W[k * (H * C) + h * C + c] * a[h * C + c];
  out[t] = s;
}

// cnt[dst[e]] += 1  (same for both layers)
__global__ void cnt_kernel(const int* __restrict__ dst, float* __restrict__ cnt, int E) {
  int e = blockIdx.x * blockDim.x + threadIdx.x;
  if (e < E) atomicAdd(&cnt[dst[e]], 1.0f);
}

// ------------------------------------------------- per-node score + attn dots
// one wave per node: escore = exp(x.ow + ob), als[h] = x.wa_src[:,h], ald[h] likewise
__global__ __launch_bounds__(256)
void node_attn_kernel(const float* __restrict__ x, const float* __restrict__ ow,
                      const float* __restrict__ obp,
                      const float* __restrict__ was, const float* __restrict__ wad,
                      float* __restrict__ escore, float* __restrict__ als,
                      float* __restrict__ ald, int N, int H) {
  int wv   = (blockIdx.x * blockDim.x + threadIdx.x) >> 5;
  int lane = threadIdx.x & 31;
  if (wv >= N) return;
  const float* xr = x + (long)wv * DM;
  float acc[9];
  #pragma unroll
  for (int i = 0; i < 9; ++i) acc[i] = 0.0f;
  #pragma unroll
  for (int i = 0; i < 4; ++i) {
    int c = lane * 4 + i;
    float xv = xr[c];
    acc[0] += xv * ow[c];
    for (int h = 0; h < H; ++h) {
      acc[1 + h] += xv * was[c * H + h];
      acc[5 + h] += xv * wad[c * H + h];
    }
  }
  for (int off = 16; off; off >>= 1)
    #pragma unroll
    for (int i = 0; i < 9; ++i) acc[i] += __shfl_down(acc[i], off, 32);
  if (lane == 0) {
    escore[wv] = __expf(acc[0] + obp[0]);
    for (int h = 0; h < H; ++h) {
      als[wv * H + h] = acc[1 + h];
      ald[wv * H + h] = acc[5 + h];
    }
  }
}

// --------------------------------------------------- WMMA fragment A loader
// 16-bit A 16x32 layout (contiguous): lane<16 -> K=j ; lane>=16 -> K=16+j
__device__ inline v16h load_a_frag(const float* __restrict__ row, int kt, int hi) {
  int c0 = kt * 32 + hi * 16;
  v16h a;
  #pragma unroll
  for (int j = 0; j < 8; ++j) a[j]     = (_Float16)row[c0 + j];
  #pragma unroll
  for (int j = 0; j < 8; ++j) a[8 + j] = (_Float16)row[c0 + 8 + j];
  return a;
}

// --------------------------------------------------------- edge encoder GEMM
// te[e,:] = relu(edge_attr[e,:] @ ew + eb) * escore[src[e]]; loopsum[dst] += te
// 128 threads = 4 waves; each wave -> 16 edges x 32 cols (2 n-tiles x 2 k-steps)
__global__ __launch_bounds__(128)
void edge_enc_kernel(const float* __restrict__ ea, const float* __restrict__ ew,
                     const float* __restrict__ eb, const float* __restrict__ escore,
                     const int* __restrict__ src, const int* __restrict__ dst,
                     float* __restrict__ te, float* __restrict__ loopsum, int E) {
  __shared__ alignas(32) _Float16 Bs[HDIM][EDIM + 16];  // transposed [n][k]
  int tid = threadIdx.x;
  for (int idx = tid; idx < EDIM * HDIM; idx += 128) {
    int k = idx / HDIM, n = idx % HDIM;
    Bs[n][k] = (_Float16)ew[idx];
  }
  __syncthreads();
  int wave = tid >> 5, lane = tid & 31;
  long ebase = ((long)blockIdx.x * 4 + wave) * 16;
  if (ebase >= E) return;
  int mrow = lane & 15;
  int hi   = (lane >> 4) & 1;
  int mA = (int)ebase + mrow;
  if (mA >= E) mA = 0;                       // clamp; OOB rows never stored
  const float* row = ea + (long)mA * EDIM;
  v16h A[2];
  #pragma unroll
  for (int kt = 0; kt < 2; ++kt) A[kt] = load_a_frag(row, kt, hi);

  // hoisted per-row metadata (rows handled by this lane, same for both n-tiles)
  bool  okr[8];
  float esc[8];
  long  dof[8];
  #pragma unroll
  for (int r = 0; r < 8; ++r) {
    int m = (int)ebase + hi * 8 + r;
    okr[r] = (m < E);
    int mm = okr[r] ? m : 0;
    esc[r] = escore[src[mm]];
    dof[r] = (long)dst[mm] * HDIM;
  }

  #pragma unroll
  for (int nt = 0; nt < 2; ++nt) {
    int n = nt * 16 + mrow;
    v8f acc = {};
    #pragma unroll
    for (int kt = 0; kt < 2; ++kt) {
      int kk = kt * 32 + hi * 16;
      v16h b = *(const v16h*)&Bs[n][kk];
      acc = __builtin_amdgcn_wmma_f32_16x16x32_f16(false, A[kt], false, b,
                                                   (short)0, acc, false, false);
    }
    float bias = eb[n];
    #pragma unroll
    for (int r = 0; r < 8; ++r) {
      if (okr[r]) {
        int m = (int)ebase + hi * 8 + r;
        float v = acc[r] + bias;
        v = v > 0.0f ? v : 0.0f;
        v *= esc[r];
        te[(long)m * HDIM + n] = v;
        atomicAdd(&loopsum[dof[r] + n], v);
      }
    }
  }
}

// ------------------------------------------------------ node transform GEMM
// Y = X @ W  (N x 128 @ 128 x 128), 256 thr = 8 waves, 16 rows/wave.
// W tile (64 KB f32) is DMA'd into LDS by the Tensor Data Mover, then
// converted to transposed f16 for the WMMA B operand.
__global__ __launch_bounds__(256)
void gemm128_kernel(const float* __restrict__ X, const float* __restrict__ W,
                    float* __restrict__ Y, int N) {
  __shared__ alignas(16) float     Wstage[DM * DM];     // 64 KB, TDM destination
  __shared__ alignas(32) _Float16 Bs[DM][DM + 16];      // 36 KB, transposed [n][k]
  int tid = threadIdx.x;

  if (tid < 32) {  // wave 0 issues one tensor DMA: global W (128x128 f32) -> LDS
    unsigned long long ga = (unsigned long long)(const void*)W;
    unsigned int lds = (unsigned int)(unsigned long long)(const void*)&Wstage[0];
    u32x4 g0;
    g0[0] = 1u;                                            // count=1 (valid user D#)
    g0[1] = lds;                                           // lds_addr (bytes)
    g0[2] = (unsigned int)ga;                              // global_addr[31:0]
    g0[3] = (unsigned int)((ga >> 32) & 0x1FFFFFFull)      // global_addr[56:32]
            | 0x80000000u;                                 // type=2 ("image")
    i32x8 g1;
    g1[0] = 2 << 16;      // workgroup_mask=0 | data_size=2 (4 bytes)
    g1[1] = DM << 16;     // tensor_dim0[15:0] = 128
    g1[2] = DM << 16;     // tensor_dim0[31:16]=0 | tensor_dim1[15:0] = 128
    g1[3] = DM << 16;     // tensor_dim1[31:16]=0 | tile_dim0 = 128
    g1[4] = DM;           // tile_dim1 = 128 | tile_dim2 = 0
    g1[5] = DM;           // tensor_dim0_stride[31:0] = 128 elements
    g1[6] = 0;            // stride[47:32]=0 | tensor_dim1_stride[15:0]=0
    g1[7] = 0;
    i32x4 gz4 = {0, 0, 0, 0};
    i32x8 gz8 = {0, 0, 0, 0, 0, 0, 0, 0};
    __builtin_amdgcn_tensor_load_to_lds(g0, g1, gz4, gz4, gz8, 0);
    __builtin_amdgcn_s_wait_tensorcnt(0);
  }
  __syncthreads();

  for (int idx = tid; idx < DM * DM; idx += 256) {
    int k = idx >> 7, n = idx & 127;
    Bs[n][k] = (_Float16)Wstage[idx];
  }
  __syncthreads();

  int wave = tid >> 5, lane = tid & 31;
  long rbase = ((long)blockIdx.x * 8 + wave) * 16;
  if (rbase >= N) return;
  int mrow = lane & 15, hi = (lane >> 4) & 1;
  int mA = (int)rbase + mrow;
  if (mA >= N) mA = N - 1;
  const float* row = X + (long)mA * DM;
  v16h A[4];
  #pragma unroll
  for (int kt = 0; kt < 4; ++kt) A[kt] = load_a_frag(row, kt, hi);
  #pragma unroll
  for (int nt = 0; nt < 8; ++nt) {
    int n = nt * 16 + mrow;
    v8f acc = {};
    #pragma unroll
    for (int kt = 0; kt < 4; ++kt) {
      int kk = kt * 32 + hi * 16;
      v16h b = *(const v16h*)&Bs[n][kk];
      acc = __builtin_amdgcn_wmma_f32_16x16x32_f16(false, A[kt], false, b,
                                                   (short)0, acc, false, false);
    }
    #pragma unroll
    for (int r = 0; r < 8; ++r) {
      int m = (int)rbase + hi * 8 + r;
      if (m < N) Y[(long)m * DM + n] = acc[r];
    }
  }
}

// ---------------------------------------------- loop_attr = loopsum / max(cnt,1)
__global__ void loopdiv_kernel(float* __restrict__ loopsum, const float* __restrict__ cnt, int N) {
  int t = blockIdx.x * blockDim.x + threadIdx.x;
  if (t >= N * HDIM) return;
  loopsum[t] /= fmaxf(cnt[t >> 5], 1.0f);
}

// ----------------------------------------------------- alpha pass 1 (+ seg max)
__global__ void alpha1_kernel(const float* __restrict__ te, const float* __restrict__ loopattr,
                              const int* __restrict__ src, const int* __restrict__ dst,
                              const float* __restrict__ wea,   // [HDIM*H]
                              const float* __restrict__ als, const float* __restrict__ ald,
                              float* __restrict__ alpha, float* __restrict__ mx,
                              int E, int N, int H) {
  int e2 = blockIdx.x * blockDim.x + threadIdx.x;
  if (e2 >= E + N) return;
  int s, d; const float* row;
  if (e2 < E) { s = src[e2]; d = dst[e2]; row = te + (long)e2 * HDIM; }
  else        { s = d = e2 - E;           row = loopattr + (long)s * HDIM; }
  float ale[4] = {0.f, 0.f, 0.f, 0.f};
  for (int k = 0; k < HDIM; ++k) {
    float v = row[k];
    for (int h = 0; h < H; ++h) ale[h] += v * wea[k * H + h];
  }
  for (int h = 0; h < H; ++h) {
    float a = als[s * H + h] + ald[d * H + h] + ale[h];
    a = a > 0.0f ? a : 0.2f * a;                         // leaky_relu(0.2)
    alpha[(long)e2 * H + h] = a;
    atomicMaxFloat(&mx[d * H + h], a);
  }
}

// --------------------------------------------------- alpha pass 2 (exp + seg sum)
__global__ void alpha2_kernel(const int* __restrict__ dst, float* __restrict__ alpha,
                              const float* __restrict__ mx, float* __restrict__ den,
                              int E, int N, int H) {
  int e2 = blockIdx.x * blockDim.x + threadIdx.x;
  if (e2 >= E + N) return;
  int d = (e2 < E) ? dst[e2] : (e2 - E);
  for (int h = 0; h < H; ++h) {
    float ex = __expf(alpha[(long)e2 * H + h] - mx[d * H + h]);
    alpha[(long)e2 * H + h] = ex;
    atomicAdd(&den[d * H + h], ex);
  }
}

// ------------------------------------------------ aggregation: out[dst] += xs[src]*coef
// one wave per (edge or self-loop); lane handles 4 contiguous channels (same head)
__global__ __launch_bounds__(256)
void aggregate_kernel(const float* __restrict__ xs, const float* __restrict__ ex,
                      const float* __restrict__ den, const int* __restrict__ src,
                      const int* __restrict__ dst, float* __restrict__ out,
                      int E, int N, int H) {
  int wv   = (blockIdx.x * blockDim.x + threadIdx.x) >> 5;
  int lane = threadIdx.x & 31;
  if (wv >= E + N) return;
  int s, d;
  if (wv < E) { s = src[wv]; d = dst[wv]; } else { s = d = wv - E; }
  int logC = (H == 4) ? 5 : 7;
  int c0 = lane * 4;
  int h  = c0 >> logC;
  float coef = ex[(long)wv * H + h] / (den[d * H + h] + 1e-16f);
  float4 xv = *(const float4*)(xs + (long)s * DM + c0);
  float* orow = out + (long)d * DM;
  atomicAdd(&orow[c0 + 0], xv.x * coef);
  atomicAdd(&orow[c0 + 1], xv.y * coef);
  atomicAdd(&orow[c0 + 2], xv.z * coef);
  atomicAdd(&orow[c0 + 3], xv.w * coef);
}

// ----------------------------------------------------------- post layers
__global__ void post0_kernel(const float* __restrict__ gat, const float* __restrict__ b,
                             const float* __restrict__ x0, float* __restrict__ x1, int N) {
  int t = blockIdx.x * blockDim.x + threadIdx.x;
  if (t >= N * DM) return;
  float v = gat[t] + b[t & 127] + x0[t];
  x1[t] = v > 0.0f ? v : 0.0f;
}

__global__ void post1_kernel(const float* __restrict__ gat, const float* __restrict__ b,
                             const float* __restrict__ x1, const float* __restrict__ x0,
                             float* __restrict__ out, int N) {
  int t = blockIdx.x * blockDim.x + threadIdx.x;
  if (t >= N * DM) return;
  out[t] = gat[t] + b[t & 127] + x1[t] + x0[t];
}

// ---------------------------------------------------------------- host side

static void run_layer(const float* xin, const float* eattr,
                      const int* src, const int* dst,
                      const float* ew, const float* eb,
                      const float* ow, const float* obp,
                      const float* W, const float* a_s, const float* a_d,
                      const float* We, const float* a_e,
                      int N, int E, int H, int C,
                      float* te, float* xs, float* gat, float* escore,
                      float* als, float* ald, float* alphabuf, float* mx, float* den,
                      float* cnt, float* loopsum,
                      float* wasf, float* wadf, float* weaf, hipStream_t stream) {
  int E2 = E + N;
  fold_kernel<<<(DM * H + 255) / 256, 256, 0, stream>>>(W, a_s, wasf, DM, H, C);
  fold_kernel<<<(DM * H + 255) / 256, 256, 0, stream>>>(W, a_d, wadf, DM, H, C);
  fold_kernel<<<(HDIM * H + 255) / 256, 256, 0, stream>>>(We, a_e, weaf, HDIM, H, C);
  node_attn_kernel<<<(N * 32 + 255) / 256, 256, 0, stream>>>(
      xin, ow, obp, wasf, wadf, escore, als, ald, N, H);
  (void)hipMemsetAsync(loopsum, 0, (size_t)N * HDIM * sizeof(float), stream);
  edge_enc_kernel<<<(E + 63) / 64, 128, 0, stream>>>(
      eattr, ew, eb, escore, src, dst, te, loopsum, E);
  loopdiv_kernel<<<(N * HDIM + 255) / 256, 256, 0, stream>>>(loopsum, cnt, N);
  gemm128_kernel<<<(N + 127) / 128, 256, 0, stream>>>(xin, W, xs, N);
  fill_kernel<<<(N * H + 255) / 256, 256, 0, stream>>>(mx, -INFINITY, N * H);
  alpha1_kernel<<<(E2 + 255) / 256, 256, 0, stream>>>(
      te, loopsum, src, dst, weaf, als, ald, alphabuf, mx, E, N, H);
  (void)hipMemsetAsync(den, 0, (size_t)N * H * sizeof(float), stream);
  alpha2_kernel<<<(E2 + 255) / 256, 256, 0, stream>>>(dst, alphabuf, mx, den, E, N, H);
  (void)hipMemsetAsync(gat, 0, (size_t)N * DM * sizeof(float), stream);
  aggregate_kernel<<<(E2 * 32 + 255) / 256, 256, 0, stream>>>(
      xs, alphabuf, den, src, dst, gat, E, N, H);
}

extern "C" void kernel_launch(void* const* d_in, const int* in_sizes, int n_in,
                              void* d_out, int out_size, void* d_ws, size_t ws_size,
                              hipStream_t stream) {
  (void)n_in; (void)out_size; (void)ws_size;
  const float* x0    = (const float*)d_in[0];
  const int*   eidx  = (const int*)d_in[1];
  const float* eattr = (const float*)d_in[2];
  const float* ew0 = (const float*)d_in[3];  const float* eb0 = (const float*)d_in[4];
  const float* ew1 = (const float*)d_in[5];  const float* eb1 = (const float*)d_in[6];
  const float* ow0 = (const float*)d_in[7];  const float* ob0 = (const float*)d_in[8];
  const float* ow1 = (const float*)d_in[9];  const float* ob1 = (const float*)d_in[10];
  const float* g0W = (const float*)d_in[11];
  const float* g0as = (const float*)d_in[12]; const float* g0ad = (const float*)d_in[13];
  const float* g0We = (const float*)d_in[14]; const float* g0ae = (const float*)d_in[15];
  const float* g0b  = (const float*)d_in[16];
  const float* g1W = (const float*)d_in[17];
  const float* g1as = (const float*)d_in[18]; const float* g1ad = (const float*)d_in[19];
  const float* g1We = (const float*)d_in[20]; const float* g1ae = (const float*)d_in[21];
  const float* g1b  = (const float*)d_in[22];

  int N = in_sizes[0] / DM;
  int E = in_sizes[1] / 2;
  const int* src = eidx;
  const int* dst = eidx + E;

  float* ws = (float*)d_ws;
  size_t off = 0;
  auto alloc = [&](size_t n) { float* p = ws + off; off += n; return p; };
  float* te      = alloc((size_t)E * HDIM);
  float* xs      = alloc((size_t)N * DM);
  float* x1      = alloc((size_t)N * DM);
  float* gat     = alloc((size_t)N * DM);
  float* escore  = alloc((size_t)N);
  float* als     = alloc((size_t)N * 4);
  float* ald     = alloc((size_t)N * 4);
  float* alphab  = alloc((size_t)(E + N) * 4);
  float* mx      = alloc((size_t)N * 4);
  float* den     = alloc((size_t)N * 4);
  float* cnt     = alloc((size_t)N);
  float* loopsum = alloc((size_t)N * HDIM);
  float* wasf    = alloc((size_t)DM * 4);
  float* wadf    = alloc((size_t)DM * 4);
  float* weaf    = alloc((size_t)HDIM * 4);

  // in-degree counts (dst-only, identical for both layers)
  (void)hipMemsetAsync(cnt, 0, (size_t)N * sizeof(float), stream);
  cnt_kernel<<<(E + 255) / 256, 256, 0, stream>>>(dst, cnt, E);

  // ---- layer 0: H=4, C=32 ----
  run_layer(x0, eattr, src, dst, ew0, eb0, ow0, ob0,
            g0W, g0as, g0ad, g0We, g0ae, N, E, 4, 32,
            te, xs, gat, escore, als, ald, alphab, mx, den, cnt, loopsum,
            wasf, wadf, weaf, stream);
  post0_kernel<<<(N * DM + 255) / 256, 256, 0, stream>>>(gat, g0b, x0, x1, N);

  // ---- layer 1: H=1, C=128 ----
  run_layer(x1, eattr, src, dst, ew1, eb1, ow1, ob1,
            g1W, g1as, g1ad, g1We, g1ae, N, E, 1, 128,
            te, xs, gat, escore, als, ald, alphab, mx, den, cnt, loopsum,
            wasf, wadf, weaf, stream);
  post1_kernel<<<(N * DM + 255) / 256, 256, 0, stream>>>(
      gat, g1b, x1, x0, (float*)d_out, N);
}
